// EncoderBlock_62629213110406
// MI455X (gfx1250) — matrix operand order repaired and checked by reading, not compiled, not run on hardware
//
#include <hip/hip_runtime.h>
#include <hip/hip_bf16.h>

typedef __attribute__((ext_vector_type(16))) _Float16 v16h;
typedef __attribute__((ext_vector_type(8)))  _Float16 v8h;
typedef __attribute__((ext_vector_type(8)))  float    v8f;
typedef __attribute__((ext_vector_type(4)))  unsigned int u32x4;
typedef __attribute__((ext_vector_type(8)))  int      i32x8;
typedef __attribute__((ext_vector_type(4)))  int      i32x4;

#define HS   1024
#define FF   4096
#define NH   16
#define HD   64
#define SEQ  2048
#define BATCH 2
#define MTOK (BATCH*SEQ)   // 4096 tokens

#define LDS_ROW 40         // 32 data halves + 8 pad halves = 80B row stride (16B aligned, bank-spread)

// ---------------------------------------------------------------------------
// WMMA operand loaders (contiguous k order: element s of lane half h holds k = 16h + s)
// ---------------------------------------------------------------------------
static __device__ __forceinline__ v16h cat8(v8h a, v8h b) {
  return __builtin_shufflevector(a, b, 0,1,2,3,4,5,6,7,8,9,10,11,12,13,14,15);
}

// A operand: 16x32 f16 tile, row-major source at p (row stride ld halves).
static __device__ __forceinline__ v16h load_a_rm(const _Float16* p, int ld, int lane) {
  int row = lane & 15;
  int kb  = (lane >> 4) << 4;
  const _Float16* rp = p + (size_t)row * ld;
  v8h c0 = *reinterpret_cast<const v8h*>(rp + kb);
  v8h c1 = *reinterpret_cast<const v8h*>(rp + kb + 8);
  return cat8(c0, c1);
}

// B operand: 32x16 tile where B[k][n] = p[n*ld + k]  (B^T row-major at p).
static __device__ __forceinline__ v16h load_b_cm(const _Float16* p, int ld, int lane) {
  int n  = lane & 15;
  int k0 = (lane >> 4) << 4;
  return *reinterpret_cast<const v16h*>(p + (size_t)n * ld + k0);
}

// B operand from LDS panel with padded row stride LDS_ROW (two 16B ds loads).
static __device__ __forceinline__ v16h load_b_lds(const _Float16* p, int lane) {
  int n  = lane & 15;
  int k0 = (lane >> 4) << 4;
  const _Float16* rp = p + n * LDS_ROW + k0;
  v8h c0 = *reinterpret_cast<const v8h*>(rp);
  v8h c1 = *reinterpret_cast<const v8h*>(rp + 8);
  return cat8(c0, c1);
}

static __device__ __forceinline__ v8f wmma16(v16h a, v16h b, v8f c) {
  return __builtin_amdgcn_wmma_f32_16x16x32_f16(false, a, false, b, (short)0, c, false, false);
}

// ---------------------------------------------------------------------------
// Tensor Data Mover: stage a 256-row x 32-half panel of Wt[N,K] into LDS.
// D# per CDNA5 ISA: 2D tile, data_size=2B,
// LDS padding 4 DWORDs every 16 DWORDs -> 80B LDS row stride.
// ---------------------------------------------------------------------------
static __device__ __forceinline__ unsigned lds_offset_of(const void* p) {
  return (unsigned)(uintptr_t)(__attribute__((address_space(3))) const char*)p;
}

static __device__ __forceinline__ void tdm_stage_panel(const _Float16* gsrc, unsigned lds_off, int K) {
  unsigned long long ga = (unsigned long long)(uintptr_t)gsrc;
  u32x4 g0;
  g0[0] = 1u;                                            // count=1, user mode
  g0[1] = lds_off;                                       // lds_addr (bytes)
  g0[2] = (unsigned)ga;                                  // global_addr lo
  g0[3] = ((unsigned)(ga >> 32) & 0x01FFFFFFu) | (2u << 30);  // addr hi | type=2
  i32x8 g1;
  g1[0] = (int)((1u << 16)            // data_size = 2 bytes
              | (1u << 20)            // pad_enable
              | (3u << 22)            // pad_interval: 16 DWORDs (64B of data)
              | (3u << 25));          // pad_amount: 4 DWORDs (16B)
  g1[1] = (int)(((unsigned)K & 0xFFFFu) << 16);          // tensor_dim0[15:0]
  g1[2] = (int)((((unsigned)K >> 16) & 0xFFFFu)          // tensor_dim0[31:16]
              | (256u << 16));                           // tensor_dim1[15:0] = 256
  g1[3] = (int)(32u << 16);                              // tensor_dim1 hi=0 | tile_dim0=32
  g1[4] = 256;                                           // tile_dim1=256, tile_dim2=0
  g1[5] = K;                                             // tensor_dim0_stride lo (elements)
  g1[6] = 0;
  g1[7] = 0;
  i32x4 z4 = {0, 0, 0, 0};
#if __has_include(<hip/amd_detail/amd_gfx1250_TDM.h>)
  i32x8 z8 = {0, 0, 0, 0, 0, 0, 0, 0};
  __builtin_amdgcn_tensor_load_to_lds(g0, g1, z4, z4, z8, 0);
#else
  __builtin_amdgcn_tensor_load_to_lds(g0, g1, z4, z4, 0);
#endif
}

// ---------------------------------------------------------------------------
// Elementwise converters
// ---------------------------------------------------------------------------
__global__ void cast_f16_kernel(const float* __restrict__ in, _Float16* __restrict__ out, int n) {
  int i = blockIdx.x * 256 + threadIdx.x;
  if (i < n) out[i] = (_Float16)in[i];
}

__global__ void transpose_cast_kernel(const float* __restrict__ in, _Float16* __restrict__ out,
                                      int R, int C) {
  int i = blockIdx.x * 256 + threadIdx.x;
  if (i >= R * C) return;
  int r = i / C, c = i % C;
  out[(size_t)c * R + r] = (_Float16)in[i];
}

// v: [B*SEQ, HS] f16  ->  vt: [B, NH, HD, SEQ] f16  (per-head transpose)
__global__ void transpose_v_kernel(const _Float16* __restrict__ v, _Float16* __restrict__ vt) {
  int i = blockIdx.x * 256 + threadIdx.x;
  int d = i % HD;
  int h = (i / HD) % NH;
  int s = (i / HS) % SEQ;
  int b = i / (SEQ * HS);
  vt[(((size_t)b * NH + h) * HD + d) * SEQ + s] = v[i];
}

// ---------------------------------------------------------------------------
// GEMM: out[M,N] = A[M,K] @ W[K,N] + bias, W transposed (Wt[N,K] f16).
// Block = 4 waves, 32x256 output tile. The 256x32 weight panel for each
// K-step is DMA'd into LDS by the Tensor Data Mover (double buffered);
// A streams register-direct. 8 WMMAs per wave per K-step.
// MODE 0: f16 out +bias.  MODE 1: f16 out +bias +ReLU.  MODE 2: f32 out +bias.
// ---------------------------------------------------------------------------
template<int MODE>
__global__ __launch_bounds__(128) void gemm_wmma(const _Float16* __restrict__ A,
                                                 const _Float16* __restrict__ Wt,
                                                 const float* __restrict__ bias,
                                                 void* __restrict__ out,
                                                 int M, int N, int K) {
  __shared__ _Float16 Wbuf[2][256 * LDS_ROW];   // 2 x 20KB

  int lane = threadIdx.x & 31;
  int warp = threadIdx.x >> 5;
  int nblk = N >> 8;
  int m0 = (blockIdx.x / nblk) * 32;
  int nband = (blockIdx.x % nblk) * 256;        // block's 256-wide column band
  int n0 = nband + warp * 64;

  v8f zero = {0.f,0.f,0.f,0.f,0.f,0.f,0.f,0.f};
  v8f acc[2][4];
#pragma unroll
  for (int mi = 0; mi < 2; ++mi)
#pragma unroll
    for (int t = 0; t < 4; ++t) acc[mi][t] = zero;

  const _Float16* Ap0 = A + (size_t)m0 * K;
  const _Float16* Ap1 = Ap0 + (size_t)16 * K;
  const _Float16* Wband = Wt + (size_t)nband * K;

  if (warp == 0) tdm_stage_panel(Wband, lds_offset_of(&Wbuf[0][0]), K);

  for (int k0 = 0; k0 < K; k0 += 32) {
    int cur = (k0 >> 5) & 1;
    if (warp == 0) __builtin_amdgcn_s_wait_tensorcnt(0);
    __syncthreads();                             // staged panel visible to all waves
    if (warp == 0 && k0 + 32 < K)
      tdm_stage_panel(Wband + k0 + 32, lds_offset_of(&Wbuf[cur ^ 1][0]), K);

    if (k0 + 128 < K)                            // stream A cachelines ahead
      __builtin_prefetch(Ap0 + (size_t)lane * K + k0 + 128, 0, 3);

    v16h a0 = load_a_rm(Ap0 + k0, K, lane);
    v16h a1 = load_a_rm(Ap1 + k0, K, lane);
    const _Float16* wb = &Wbuf[cur][0] + warp * 64 * LDS_ROW;
#pragma unroll
    for (int t = 0; t < 4; ++t) {
      v16h b = load_b_lds(wb + t * 16 * LDS_ROW, lane);
      acc[0][t] = wmma16(a0, b, acc[0][t]);
      acc[1][t] = wmma16(a1, b, acc[1][t]);
    }
  }

  int col = lane & 15, hi = lane >> 4;
#pragma unroll
  for (int mi = 0; mi < 2; ++mi)
#pragma unroll
    for (int t = 0; t < 4; ++t) {
      float bv = bias[n0 + t * 16 + col];
#pragma unroll
      for (int r = 0; r < 8; ++r) {
        float v = acc[mi][t][r] + bv;
        if (MODE == 1) v = fmaxf(v, 0.f);
        size_t cidx = (size_t)(m0 + mi * 16 + r + (hi << 3)) * N + n0 + t * 16 + col;
        if (MODE == 2) ((float*)out)[cidx] = v;
        else           ((_Float16*)out)[cidx] = (_Float16)v;
      }
    }
}

// ---------------------------------------------------------------------------
// Flash attention: one wave per (16 query rows, head, batch). KV streamed in
// tiles of 32. P goes C-layout -> LDS -> A-layout for the PV WMMA.
// ---------------------------------------------------------------------------
__global__ __launch_bounds__(32) void attn_kernel(const _Float16* __restrict__ q16,
                                                  const _Float16* __restrict__ k16,
                                                  const _Float16* __restrict__ vt16,
                                                  const float* __restrict__ mask,
                                                  _Float16* __restrict__ attn16) {
  int lane = threadIdx.x;
  int m0 = blockIdx.x * 16;
  int h  = blockIdx.y;
  int b  = blockIdx.z;
  int col = lane & 15, hi = lane >> 4;

  const _Float16* qb = q16 + ((size_t)(b * SEQ + m0)) * HS + h * HD;
  v16h qa0 = load_a_rm(qb,      HS, lane);
  v16h qa1 = load_a_rm(qb + 32, HS, lane);

  const _Float16* vtb = vt16 + (((size_t)b * NH + h) * HD) * SEQ;  // [HD, SEQ]

  v8f zero = {0.f,0.f,0.f,0.f,0.f,0.f,0.f,0.f};
  v8f o[4];
  float Mr[8], Lr[8];
#pragma unroll
  for (int t = 0; t < 4; ++t) o[t] = zero;
#pragma unroll
  for (int r = 0; r < 8; ++r) { Mr[r] = -3.0e38f; Lr[r] = 0.f; }

  __shared__ _Float16 Plds[16 * 32];
  const float* mrow = mask + (size_t)b * SEQ;

  for (int kv0 = 0; kv0 < SEQ; kv0 += 32) {
    const _Float16* kb0 = k16 + ((size_t)(b * SEQ + kv0)) * HS + h * HD;
    const _Float16* kb1 = kb0 + (size_t)16 * HS;
    if (kv0 + 32 < SEQ) {
      __builtin_prefetch(kb0 + (size_t)(32 + lane) * HS, 0, 3);
      __builtin_prefetch(vtb + (size_t)lane * SEQ + kv0 + 32, 0, 3);
      __builtin_prefetch(vtb + (size_t)(32 + lane) * SEQ + kv0 + 32, 0, 3);
    }
    v8f s0 = zero, s1 = zero;
    s0 = wmma16(qa0, load_b_cm(kb0,      HS, lane), s0);
    s0 = wmma16(qa1, load_b_cm(kb0 + 32, HS, lane), s0);
    s1 = wmma16(qa0, load_b_cm(kb1,      HS, lane), s1);
    s1 = wmma16(qa1, load_b_cm(kb1 + 32, HS, lane), s1);

    float mk0 = mrow[kv0 + col];
    float mk1 = mrow[kv0 + 16 + col];
    float alpha[8];
#pragma unroll
    for (int r = 0; r < 8; ++r) {
      float a0 = s0[r] * 0.125f + mk0;
      float a1 = s1[r] * 0.125f + mk1;
      float t = fmaxf(a0, a1);
      t = fmaxf(t, __shfl_xor(t, 1, 32));
      t = fmaxf(t, __shfl_xor(t, 2, 32));
      t = fmaxf(t, __shfl_xor(t, 4, 32));
      t = fmaxf(t, __shfl_xor(t, 8, 32));
      float mn = fmaxf(Mr[r], t);
      alpha[r] = __expf(Mr[r] - mn);
      Mr[r] = mn;
      a0 = __expf(a0 - mn);
      a1 = __expf(a1 - mn);
      s0[r] = a0; s1[r] = a1;
      float ss = a0 + a1;
      ss += __shfl_xor(ss, 1, 32);
      ss += __shfl_xor(ss, 2, 32);
      ss += __shfl_xor(ss, 4, 32);
      ss += __shfl_xor(ss, 8, 32);
      Lr[r] = Lr[r] * alpha[r] + ss;
    }

    __syncthreads();
#pragma unroll
    for (int r = 0; r < 8; ++r) {
      int row = r + (hi << 3);
      Plds[row * 32 + col]      = (_Float16)s0[r];
      Plds[row * 32 + 16 + col] = (_Float16)s1[r];
    }
    __syncthreads();
    v16h pa = load_a_rm(Plds, 32, lane);

#pragma unroll
    for (int t = 0; t < 4; ++t) {
#pragma unroll
      for (int r = 0; r < 8; ++r) o[t][r] *= alpha[r];
      v16h vb = load_b_cm(vtb + (size_t)(t * 16) * SEQ + kv0, SEQ, lane);
      o[t] = wmma16(pa, vb, o[t]);
    }
  }

#pragma unroll
  for (int t = 0; t < 4; ++t)
#pragma unroll
    for (int r = 0; r < 8; ++r) {
      int row = m0 + r + (hi << 3);
      attn16[((size_t)(b * SEQ + row)) * HS + h * HD + t * 16 + col] =
          (_Float16)(o[t][r] / Lr[r]);
    }
}

// ---------------------------------------------------------------------------
// Fused residual add + LayerNorm (one 256-thread block per row of 1024)
// ---------------------------------------------------------------------------
template<bool WRITE_F16>
__global__ __launch_bounds__(256) void add_ln_kernel(const float* __restrict__ x,
                                                     const float* __restrict__ delta,
                                                     const float* __restrict__ g,
                                                     const float* __restrict__ bta,
                                                     float* __restrict__ out32,
                                                     _Float16* __restrict__ out16) {
  int row = blockIdx.x;
  int tid = threadIdx.x;
  const float* xr = x     + (size_t)row * HS;
  const float* dr = delta + (size_t)row * HS;
  float vals[4];
  float s = 0.f;
#pragma unroll
  for (int i = 0; i < 4; ++i) { vals[i] = xr[tid + i*256] + dr[tid + i*256]; s += vals[i]; }

  __shared__ float red[256];
  red[tid] = s; __syncthreads();
  for (int off = 128; off > 0; off >>= 1) {
    if (tid < off) red[tid] += red[tid + off];
    __syncthreads();
  }
  float mean = red[0] * (1.f / HS);
  __syncthreads();

  float vs = 0.f;
#pragma unroll
  for (int i = 0; i < 4; ++i) { float d = vals[i] - mean; vs += d * d; }
  red[tid] = vs; __syncthreads();
  for (int off = 128; off > 0; off >>= 1) {
    if (tid < off) red[tid] += red[tid + off];
    __syncthreads();
  }
  float rstd = rsqrtf(red[0] * (1.f / HS) + 1e-5f);

#pragma unroll
  for (int i = 0; i < 4; ++i) {
    int c = tid + i * 256;
    float y = (vals[i] - mean) * rstd * g[c] + bta[c];
    out32[(size_t)row * HS + c] = y;
    if (WRITE_F16) out16[(size_t)row * HS + c] = (_Float16)y;
  }
}

// ---------------------------------------------------------------------------
// Host-side orchestration
// ---------------------------------------------------------------------------
extern "C" void kernel_launch(void* const* d_in, const int* in_sizes, int n_in,
                              void* d_out, int out_size, void* d_ws, size_t ws_size,
                              hipStream_t stream) {
  (void)in_sizes; (void)n_in; (void)out_size; (void)ws_size;
  const float* x    = (const float*)d_in[0];
  const float* mask = (const float*)d_in[1];
  const float* Wq = (const float*)d_in[2];  const float* bq = (const float*)d_in[3];
  const float* Wk = (const float*)d_in[4];  const float* bk = (const float*)d_in[5];
  const float* Wv = (const float*)d_in[6];  const float* bv = (const float*)d_in[7];
  const float* Wo = (const float*)d_in[8];  const float* bo = (const float*)d_in[9];
  const float* W1 = (const float*)d_in[10]; const float* b1 = (const float*)d_in[11];
  const float* W2 = (const float*)d_in[12]; const float* b2 = (const float*)d_in[13];
  const float* ln1g = (const float*)d_in[14]; const float* ln1b = (const float*)d_in[15];
  const float* ln2g = (const float*)d_in[16]; const float* ln2b = (const float*)d_in[17];

  char* ws = (char*)d_ws;
  size_t off = 0;
  auto alloc = [&](size_t bytes) { char* p = ws + off; off += (bytes + 255) & ~(size_t)255; return p; };
  _Float16* x16   = (_Float16*)alloc((size_t)MTOK * HS * 2);
  _Float16* WqT   = (_Float16*)alloc((size_t)HS * HS * 2);
  _Float16* WkT   = (_Float16*)alloc((size_t)HS * HS * 2);
  _Float16* WvT   = (_Float16*)alloc((size_t)HS * HS * 2);
  _Float16* WoT   = (_Float16*)alloc((size_t)HS * HS * 2);
  _Float16* W1T   = (_Float16*)alloc((size_t)HS * FF * 2);   // [FF,HS]
  _Float16* W2T   = (_Float16*)alloc((size_t)FF * HS * 2);   // [HS,FF]
  _Float16* q16   = (_Float16*)alloc((size_t)MTOK * HS * 2);
  _Float16* k16   = (_Float16*)alloc((size_t)MTOK * HS * 2);
  _Float16* v16b  = (_Float16*)alloc((size_t)MTOK * HS * 2);
  _Float16* vt16  = (_Float16*)alloc((size_t)MTOK * HS * 2); // [B,NH,HD,SEQ]
  _Float16* attn16= (_Float16*)alloc((size_t)MTOK * HS * 2);
  float*    tmp32 = (float*)   alloc((size_t)MTOK * HS * 4);
  float*    h32   = (float*)   alloc((size_t)MTOK * HS * 4);
  _Float16* h16   = (_Float16*)alloc((size_t)MTOK * HS * 2);
  _Float16* ff16  = (_Float16*)alloc((size_t)MTOK * FF * 2);

  cast_f16_kernel<<<(MTOK * HS) / 256, 256, 0, stream>>>(x, x16, MTOK * HS);
  transpose_cast_kernel<<<(HS * HS) / 256, 256, 0, stream>>>(Wq, WqT, HS, HS);
  transpose_cast_kernel<<<(HS * HS) / 256, 256, 0, stream>>>(Wk, WkT, HS, HS);
  transpose_cast_kernel<<<(HS * HS) / 256, 256, 0, stream>>>(Wv, WvT, HS, HS);
  transpose_cast_kernel<<<(HS * HS) / 256, 256, 0, stream>>>(Wo, WoT, HS, HS);
  transpose_cast_kernel<<<(HS * FF) / 256, 256, 0, stream>>>(W1, W1T, HS, FF);
  transpose_cast_kernel<<<(FF * HS) / 256, 256, 0, stream>>>(W2, W2T, FF, HS);

  const int gproj = (MTOK / 32) * (HS / 256);   // 512 blocks
  const int gff1  = (MTOK / 32) * (FF / 256);   // 2048 blocks

  gemm_wmma<0><<<gproj, 128, 0, stream>>>(x16, WqT, bq, q16, MTOK, HS, HS);
  gemm_wmma<0><<<gproj, 128, 0, stream>>>(x16, WkT, bk, k16, MTOK, HS, HS);
  gemm_wmma<0><<<gproj, 128, 0, stream>>>(x16, WvT, bv, v16b, MTOK, HS, HS);
  transpose_v_kernel<<<(MTOK * HS) / 256, 256, 0, stream>>>(v16b, vt16);

  attn_kernel<<<dim3(SEQ / 16, NH, BATCH), 32, 0, stream>>>(q16, k16, vt16, mask, attn16);

  gemm_wmma<2><<<gproj, 128, 0, stream>>>(attn16, WoT, bo, tmp32, MTOK, HS, HS);
  add_ln_kernel<true><<<MTOK, 256, 0, stream>>>(x, tmp32, ln1g, ln1b, h32, h16);

  gemm_wmma<1><<<gff1, 128, 0, stream>>>(h16, W1T, b1, ff16, MTOK, FF, HS);
  gemm_wmma<2><<<gproj, 128, 0, stream>>>(ff16, W2T, b2, tmp32, MTOK, HS, FF);
  add_ln_kernel<false><<<MTOK, 256, 0, stream>>>(h32, tmp32, ln2g, ln2b, (float*)d_out, nullptr);
}
